// VSSM_27427661152994
// MI455X (gfx1250) — matrix operand order repaired and checked by reading, not compiled, not run on hardware
//
#include <hip/hip_runtime.h>
#include <hip/hip_bf16.h>
#include <math.h>

typedef __attribute__((ext_vector_type(16))) _Float16 v16h;
typedef __attribute__((ext_vector_type(8)))  float    v8f;
typedef __attribute__((ext_vector_type(4)))  unsigned int v4u;
typedef __attribute__((ext_vector_type(8)))  int      v8i;
typedef __attribute__((ext_vector_type(4)))  int      v4i;

#define D_MODEL 96
#define D_INNER 192
#define D_STATE 16
#define DT_RANK 6
#define KDIRS   4
#define BSZ     8
#define HSZ     56
#define WSZ     56
#define LSEQ    (HSZ * WSZ)   // 3136
#define MTOT    (BSZ * LSEQ)  // 25088
#define SCAN_CH 64            // scan chunk length (steps per TDM transfer)
#define NCHUNK  (LSEQ / SCAN_CH)  // 49

// Cross-scan traversal mapping: flat sequence index l (for direction dir)
// -> spatial (h,w) inside one image.
__device__ __forceinline__ void map_dir(int dir, int l, int& hh, int& ww) {
    int l2 = (dir >= 2) ? (LSEQ - 1 - l) : l;
    int q = l2 / WSZ, r = l2 % WSZ;  // H == W == 56
    if (dir & 1) { ww = q; hh = r; } else { hh = q; ww = r; }
}

// ---------------------------------------------------------------------------
// WMMA GEMM: out[M,N] = A[M,K] * W[N,K]^T  (f16 inputs, f32 accumulate)
// One wave per 16(M) x 64(N) super-tile: A fragment loaded once per K-chunk,
// reused across 4 v_wmma_f32_16x16x32_f16 issues (4 accumulators).
//   mode 0: in_proj   — epilogue splits xi / silu(z)
//   mode 1: x_dbl     — A rows gathered via cross-scan mapping
//   mode 2: out_proj  — plain store with row stride ldOut
// ---------------------------------------------------------------------------
__global__ __launch_bounds__(32) void wmma_gemm(
    const float* __restrict__ A, const float* __restrict__ W,
    float* __restrict__ out0, float* __restrict__ out1,
    int K, int Nw, int ldOut, int mode, int dir)
{
    const int lane = threadIdx.x;
    const int bm = blockIdx.x;
    const int bn0 = blockIdx.y * 4;
    const int mrow = bm * 16 + (lane & 15);

    // A-row base pointer (cross-scan gather for mode 1)
    const float* arow;
    if (mode == 1) {
        int b = mrow / LSEQ, l = mrow % LSEQ, hh, ww;
        map_dir(dir, l, hh, ww);
        arow = A + (size_t)((b * HSZ + hh) * WSZ + ww) * D_INNER;
    } else {
        arow = A + (size_t)mrow * K;
    }

    const float* wrow[4];
    int ncol[4];
    #pragma unroll
    for (int t = 0; t < 4; ++t) {
        ncol[t] = (bn0 + t) * 16 + (lane & 15);
        wrow[t] = (ncol[t] < Nw) ? (W + (size_t)ncol[t] * K) : nullptr;
    }

    // ISA 7.12.2 fragment K-offsets per lane half
    const int kaoff = (lane & 16) ? 8 : 0;   // A: 16x32 f16
    const int kboff = (lane & 16) ? 8 : 0;   // B: 32x16 f16

    v8f acc[4] = {{}, {}, {}, {}};
    for (int kb = 0; kb < K; kb += 32) {
        v16h af;
        // A fragment: halves 0..7 -> K=kaoff+e ; halves 8..15 -> K=kaoff+e+8
        #pragma unroll
        for (int e = 0; e < 16; e += 2) {
            int k = kb + kaoff + e + ((e >= 8) ? 8 : 0);
            float2 v = *(const float2*)(arow + k);
            af[e]     = (_Float16)v.x;
            af[e + 1] = (_Float16)v.y;
        }
        #pragma unroll
        for (int t = 0; t < 4; ++t) {
            v16h bf;
            if (wrow[t]) {
                #pragma unroll
                for (int h4 = 0; h4 < 16; h4 += 4) {
                    float4 v = *(const float4*)(wrow[t] + kb + kboff + h4 + ((h4 >= 8) ? 8 : 0));
                    bf[h4]     = (_Float16)v.x;
                    bf[h4 + 1] = (_Float16)v.y;
                    bf[h4 + 2] = (_Float16)v.z;
                    bf[h4 + 3] = (_Float16)v.w;
                }
            } else {
                #pragma unroll
                for (int h4 = 0; h4 < 16; ++h4) bf[h4] = (_Float16)0.f;
            }
            acc[t] = __builtin_amdgcn_wmma_f32_16x16x32_f16(
                false, af, false, bf, (short)0, acc[t], false, false);
        }
    }

    // Epilogue: C/D layout — VGPR r holds (m = r + 8*(lane/16), n = lane%16)
    #pragma unroll
    for (int t = 0; t < 4; ++t) {
        #pragma unroll
        for (int r = 0; r < 8; ++r) {
            int m = bm * 16 + r + ((lane >> 4) << 3);
            int n = ncol[t];
            float v = acc[t][r];
            if (mode == 0) {
                if (n < D_INNER)
                    out0[(size_t)m * D_INNER + n] = v;                      // xi
                else
                    out1[(size_t)m * D_INNER + (n - D_INNER)] =
                        v / (1.f + __expf(-v));                             // silu(z)
            } else {
                if (n < Nw) out0[(size_t)m * ldOut + n] = v;
            }
        }
    }
}

// ---------------------------------------------------------------------------
// Depthwise 3x3 SAME conv + bias + SiLU (NHWC, one thread per (pixel,chan))
// ---------------------------------------------------------------------------
__global__ __launch_bounds__(192) void dwconv_silu(
    const float* __restrict__ xi, const float* __restrict__ cw,
    const float* __restrict__ cb, float* __restrict__ xc)
{
    int p = blockIdx.x;           // b*H*W + h*W + w
    int c = threadIdx.x;
    int w = p % WSZ, h = (p / WSZ) % HSZ, b = p / (HSZ * WSZ);
    float acc = cb[c];
    #pragma unroll
    for (int dy = 0; dy < 3; ++dy) {
        int hy = h + dy - 1;
        if (hy < 0 || hy >= HSZ) continue;
        #pragma unroll
        for (int dx = 0; dx < 3; ++dx) {
            int wx = w + dx - 1;
            if (wx < 0 || wx >= WSZ) continue;
            acc += xi[((size_t)(b * HSZ + hy) * WSZ + wx) * D_INNER + c] *
                   cw[c * 9 + dy * 3 + dx];
        }
    }
    xc[(size_t)p * D_INNER + c] = acc / (1.f + __expf(-acc));
}

// ---------------------------------------------------------------------------
// TDM issue: 1-D contiguous copy of SCAN_CH*48 f32 (12 KB) global -> LDS.
// D# packing per ISA 8.3/8.4: count=1, type=2, data_size=4B, tile_dim0=3072.
// This toolchain declares the 6-arg builtin:
//   (uint32x4 g0, int32x8 g1, int32x4, int32x4, int32x8, i32 cpol)
// ---------------------------------------------------------------------------
__device__ __forceinline__ void tdm_load_chunk(const float* gsrc, uint32_t lds_byte)
{
    const uint32_t nelem = SCAN_CH * 48;             // 3072 elements
    uint64_t ga = (uint64_t)(uintptr_t)gsrc;
    v4u g0;
    g0.x = 0x1u;                                     // count=1, user mode
    g0.y = lds_byte;                                 // lds_addr [63:32]
    g0.z = (uint32_t)ga;                             // global_addr[31:0]
    g0.w = (uint32_t)((ga >> 32) & 0x1FFFFFFu)       // global_addr[56:32]
         | 0x80000000u;                              // type=2 at [127:126]
    v8i g1;
    g1[0] = (int)(0x2u << 16);                       // data_size=4B; no mask
    g1[1] = (int)((nelem & 0xFFFFu) << 16);          // tensor_dim0[15:0]
    g1[2] = (int)(((nelem >> 16) & 0xFFFFu)          // tensor_dim0[31:16]
         | (1u << 16));                              // tensor_dim1 = 1
    g1[3] = (int)((nelem & 0xFFFFu) << 16);          // tile_dim0 = 3072
    g1[4] = 0;                                       // tile_dim1/2 unused
    g1[5] = (int)nelem;                              // tensor_dim0_stride
    g1[6] = 0;
    g1[7] = 0;
    v4i gz4 = {0, 0, 0, 0};
    v8i gz8 = {0, 0, 0, 0, 0, 0, 0, 0};
    __builtin_amdgcn_tensor_load_to_lds(g0, g1, gz4, gz4, gz8, 0);
}

// ---------------------------------------------------------------------------
// Selective scan: one block per (b,k); thread d carries 16-state h in regs.
// x_dbl stream (rows of 48 f32) double-buffered through LDS via TDM in
// 64-step chunks; barriers only at chunk granularity.
// ---------------------------------------------------------------------------
__global__ __launch_bounds__(192) void selective_scan(
    const float* __restrict__ xc, const float* __restrict__ xdbl,
    const float* __restrict__ Alog, const float* __restrict__ dtw,
    const float* __restrict__ dtb, const float* __restrict__ Dsk,
    float* __restrict__ ybuf)
{
    const int bk = blockIdx.x;
    const int b = bk / KDIRS, k = bk % KDIRS;
    const int d = threadIdx.x;
    const int kd = k * D_INNER + d;

    float Arow[D_STATE];
    #pragma unroll
    for (int n = 0; n < D_STATE; ++n) Arow[n] = -__expf(Alog[kd * D_STATE + n]);
    float wdt[DT_RANK];
    #pragma unroll
    for (int r = 0; r < DT_RANK; ++r) wdt[r] = dtw[kd * DT_RANK + r];
    const float bias  = dtb[kd];
    const float dskip = Dsk[kd];

    float hst[D_STATE];
    #pragma unroll
    for (int n = 0; n < D_STATE; ++n) hst[n] = 0.f;

    __shared__ float xbuf[2][SCAN_CH * 48];          // 24 KB double buffer
    const float* xrow = xdbl + (size_t)(k * MTOT + b * LSEQ) * 48;
    float* yrow = ybuf + ((size_t)(k * BSZ + b) * D_INNER + d) * LSEQ;

    // Prime buffer 0 (TDM is per-wave; only wave 0 issues)
    if (threadIdx.x < 32)
        tdm_load_chunk(xrow, (uint32_t)(uintptr_t)&xbuf[0][0]);

    for (int c = 0; c < NCHUNK; ++c) {
        if (c + 1 < NCHUNK) {
            if (threadIdx.x < 32) {
                tdm_load_chunk(xrow + (size_t)(c + 1) * SCAN_CH * 48,
                               (uint32_t)(uintptr_t)&xbuf[(c + 1) & 1][0]);
                __builtin_amdgcn_s_wait_tensorcnt(1);  // chunk c complete
            }
        } else {
            if (threadIdx.x < 32)
                __builtin_amdgcn_s_wait_tensorcnt(0);  // last chunk complete
        }
        __syncthreads();

        const float* xb = &xbuf[c & 1][0];
        for (int i = 0; i < SCAN_CH; ++i) {
            const int l = c * SCAN_CH + i;
            const float* sh = xb + i * 48;           // [0:6)=dt [6:22)=B [22:38)=C

            int hh, ww;
            map_dir(k, l, hh, ww);
            float u = xc[((size_t)(b * HSZ + hh) * WSZ + ww) * D_INNER + d];

            float dv = bias;
            #pragma unroll
            for (int r = 0; r < DT_RANK; ++r) dv += sh[r] * wdt[r];
            float delta = (dv > 20.f) ? dv : log1pf(__expf(dv));  // softplus
            float du = delta * u;

            float y = 0.f;
            #pragma unroll
            for (int n = 0; n < D_STATE; ++n) {
                hst[n] = __expf(delta * Arow[n]) * hst[n] + du * sh[6 + n];
                y += hst[n] * sh[22 + n];
            }
            yrow[l] = y + dskip * u;
        }
        __syncthreads();   // all waves done with xbuf[c&1] before reuse
    }
}

// ---------------------------------------------------------------------------
// Cross-merge (undo flips / transpose, sum 4 dirs) + LayerNorm + gate
// ---------------------------------------------------------------------------
__global__ __launch_bounds__(192) void merge_ln_gate(
    const float* __restrict__ ybuf, const float* __restrict__ zb,
    const float* __restrict__ gamma, const float* __restrict__ beta,
    float* __restrict__ yg)
{
    int bl = blockIdx.x;
    int b = bl / LSEQ, l = bl % LSEQ;
    int d = threadIdx.x;
    int h = l / WSZ, w = l % WSZ;
    int lt = w * WSZ + h;  // transposed traversal index (H == W)

    auto yat = [&](int k, int ll) {
        return ybuf[((size_t)(k * BSZ + b) * D_INNER + d) * LSEQ + ll];
    };
    float v = yat(0, l) + yat(2, LSEQ - 1 - l) +
              yat(1, lt) + yat(3, LSEQ - 1 - lt);

    __shared__ float s1[192], s2[192];
    s1[d] = v; s2[d] = v * v;
    __syncthreads();
    if (d < 64) {
        s1[d] += s1[d + 64] + s1[d + 128];
        s2[d] += s2[d + 64] + s2[d + 128];
    }
    __syncthreads();
    for (int off = 32; off >= 1; off >>= 1) {
        if (d < off) { s1[d] += s1[d + off]; s2[d] += s2[d + off]; }
        __syncthreads();
    }
    float mu  = s1[0] * (1.f / 192.f);
    float var = s2[0] * (1.f / 192.f) - mu * mu;
    float yn = (v - mu) * rsqrtf(var + 1e-5f) * gamma[d] + beta[d];
    yg[(size_t)bl * D_INNER + d] = yn * zb[(size_t)bl * D_INNER + d];
}

// ---------------------------------------------------------------------------
extern "C" void kernel_launch(void* const* d_in, const int* in_sizes, int n_in,
                              void* d_out, int out_size, void* d_ws, size_t ws_size,
                              hipStream_t stream)
{
    (void)in_sizes; (void)n_in; (void)out_size; (void)ws_size;
    const float* x    = (const float*)d_in[0];
    const float* Win  = (const float*)d_in[1];   // (384, 96)
    const float* cw   = (const float*)d_in[2];   // (192,1,3,3)
    const float* cb   = (const float*)d_in[3];   // (192)
    const float* xpw  = (const float*)d_in[4];   // (4, 38, 192)
    const float* dtw  = (const float*)d_in[5];   // (4, 192, 6)
    const float* dtb  = (const float*)d_in[6];   // (4, 192)
    const float* Alog = (const float*)d_in[7];   // (768, 16)
    const float* Dsk  = (const float*)d_in[8];   // (768)
    const float* gam  = (const float*)d_in[9];
    const float* bet  = (const float*)d_in[10];
    const float* Wout = (const float*)d_in[11];  // (96, 192)
    float* out = (float*)d_out;

    // Workspace layout (floats); total ~154 MB, all L2-resident.
    float* ws   = (float*)d_ws;
    float* xi   = ws;                              // M*192
    float* zb   = xi   + (size_t)MTOT * D_INNER;   // M*192  (silu(z))
    float* xc   = zb   + (size_t)MTOT * D_INNER;   // M*192  (conv out)
    float* xdbl = xc   + (size_t)MTOT * D_INNER;   // 4*M*48 = M*192
    float* ybuf = xdbl + (size_t)MTOT * 192;       // 4*M*192
    float* yg   = xi;                              // reuse xi after conv

    // 1. in_proj:  (25088,96) x (384,96)^T -> xi | silu(z)
    wmma_gemm<<<dim3(MTOT / 16, (2 * D_INNER) / 64), 32, 0, stream>>>(
        x, Win, xi, zb, D_MODEL, 2 * D_INNER, D_INNER, 0, 0);

    // 2. depthwise conv 3x3 + SiLU
    dwconv_silu<<<dim3(MTOT), 192, 0, stream>>>(xi, cw, cb, xc);

    // 3. x_dbl per direction: gathered (25088,192) x (38,192)^T -> (25088,48)
    for (int k = 0; k < KDIRS; ++k)
        wmma_gemm<<<dim3(MTOT / 16, 1), 32, 0, stream>>>(
            xc, xpw + (size_t)k * 38 * D_INNER,
            xdbl + (size_t)k * MTOT * 48, nullptr,
            D_INNER, 38, 48, 1, k);

    // 4. selective scan (sequential over L, parallel over B*K*D; TDM-fed)
    selective_scan<<<dim3(BSZ * KDIRS), 192, 0, stream>>>(
        xc, xdbl, Alog, dtw, dtb, Dsk, ybuf);

    // 5. cross-merge + LayerNorm + gate
    merge_ln_gate<<<dim3(MTOT), 192, 0, stream>>>(ybuf, zb, gam, bet, yg);

    // 6. out_proj: (25088,192) x (96,192)^T -> d_out (25088,96)
    wmma_gemm<<<dim3(MTOT / 16, 2), 32, 0, stream>>>(
        yg, Wout, out, nullptr, D_INNER, D_MODEL, D_MODEL, 2, 0);
}
